// MWL_Transformer_54039278518546
// MI455X (gfx1250) — matrix operand order repaired and checked by reading, not compiled, not run on hardware
//
#include <hip/hip_runtime.h>
#include <hip/hip_bf16.h>
#include <math.h>

// ---------------------------------------------------------------------------
// MI455X / gfx1250 implementation.  wave32, WMMA f32_16x16x32_f16 everywhere.
// Activations are feature-major [C, B*L] through the whole net so conv im2col
// GEMMs, transformer GEMMs (W[O,I] x X[I,N]) and attention share one layout.
// GEMM: A tile moved by the Tensor Data Mover (one tensor_load_to_lds D# per
// K-step, TENSORcnt) with an async global->LDS fallback; B tile staged
// transposed so all WMMA fragment reads are two ds_load_b128.  N is a
// compile-time constant (NTOK) so B column-read offsets fold into immediates.
// ---------------------------------------------------------------------------

#if defined(__has_builtin)
#if __has_builtin(__builtin_amdgcn_tensor_load_to_lds)
#define HAVE_TDM 1
#endif
#endif
#ifndef HAVE_TDM
#define HAVE_TDM 0
#endif

typedef __attribute__((ext_vector_type(16))) _Float16     v16h;
typedef __attribute__((ext_vector_type(8)))  _Float16     v8h;
typedef __attribute__((ext_vector_type(8)))  float        v8f;
typedef __attribute__((ext_vector_type(4)))  unsigned int v4u;
typedef __attribute__((ext_vector_type(8)))  int          v8i_;
typedef __attribute__((ext_vector_type(4)))  int          v4i_;

#define B_    2
#define L_    2048
#define F_    256
#define D_    256
#define H_    8
#define R_    4
#define KW    11
#define PAD_  5
#define DH_   32
#define NTOK  4096              // B_ * L_   (GEMM N dimension, fixed)
#define CKK   2816              // F_ * KW (conv reduction dim)
#define SOFT_SCALE 0.17677669529663689f  // 1/sqrt(32)

// ---------------------------------------------------------------------------
// WMMA helpers (CDNA5 VGPR layouts per ISA 7.12.2)
// ---------------------------------------------------------------------------
__device__ __forceinline__ v8f wmma32(v16h a, v16h b, v8f c) {
  return __builtin_amdgcn_wmma_f32_16x16x32_f16(false, a, false, b, (short)0, c,
                                                false, false);
}

__device__ __forceinline__ v8f zero_v8f() {
  v8f z;
#pragma unroll
  for (int r = 0; r < 8; ++r) z[r] = 0.0f;
  return z;
}

__device__ __forceinline__ v16h concat8(v8h lo, v8h hi) {
  return __builtin_shufflevector(lo, hi, 0, 1, 2, 3, 4, 5, 6, 7, 8, 9, 10, 11,
                                 12, 13, 14, 15);
}

// A fragment (16x32 MxK) from a row-major [*,32] tile (LDS or global).
// lanes 0-15: M=lane, K 0-15; lanes 16-31: M=lane-16, K 16-31
// Per lane: two contiguous 16B chunks -> two b128 loads.
__device__ __forceinline__ v16h load_a_frag32(const _Float16* A, int row0,
                                              int lane) {
  const int m  = row0 + (lane & 15);
  const int kb = ((lane >> 4) & 1) * 16;
  const v8h* p = (const v8h*)(A + m * 32 + kb);
  return concat8(p[0], p[1]);  // halves [kb..kb+15]
}

// B fragment (32x16 KxN) from an N-major tile BsT[n][kk] with 32-half rows.
// lanes 0-15: K=0..15, N=lane; lanes 16-31: K=16..31, N=lane-16.
// Per lane: 32 contiguous bytes -> two b128 loads.
__device__ __forceinline__ v16h load_b_frag32T(const _Float16* BsT, int col0,
                                               int lane) {
  const int n  = col0 + (lane & 15);
  const int kb = ((lane >> 4) & 1) * 16;
  const v8h* p = (const v8h*)(BsT + n * 32 + kb);
  return concat8(p[0], p[1]);
}

// B fragment from an N-major global source with arbitrary leading dim:
// B[k][n] = Bt[(n0+n)*ld + k]  (per lane contiguous -> two b128 loads)
__device__ __forceinline__ v16h load_b_fragT_g(const _Float16* Bt, int ld,
                                               int n0, int lane) {
  const int n  = lane & 15;
  const int kb = ((lane >> 4) & 1) * 16;
  const v8h* p = (const v8h*)(Bt + (size_t)(n0 + n) * ld + kb);
  return concat8(p[0], p[1]);
}

// A fragment from a K-major (transposed) source: A[m][k] = At[k*ld + m].
// Strided per lane (used once per wave for Q; loads are wave-coalesced).
__device__ __forceinline__ v16h load_a_fragT_g(const _Float16* At, int ld,
                                               int lane) {
  v16h f;
  const int m  = lane & 15;
  const int kb = ((lane >> 4) & 1) * 16;
#pragma unroll
  for (int j = 0; j < 8; ++j)  f[j] = At[(size_t)(kb + j) * ld + m];
#pragma unroll
  for (int j = 8; j < 16; ++j) f[j] = At[(size_t)(kb + j) * ld + m];
  return f;
}

// B fragment from a K-major global source: B[k][n] = Bp[(k0+k)*ldb + col0+n].
// Strided per lane but coalesced across the wave (keys dimension); ldb is a
// compile-time constant at every call site so offsets fold into immediates.
__device__ __forceinline__ v16h load_b_frag_g(const _Float16* Bp, int ldb,
                                              int col0, int lane) {
  v16h f;
  const int n  = col0 + (lane & 15);
  const int kb = ((lane >> 4) & 1) * 16;
  const _Float16* p = Bp + (size_t)kb * ldb + n;
#pragma unroll
  for (int j = 0; j < 16; ++j) f[j] = p[(size_t)j * ldb];
  return f;
}

// ---------------------------------------------------------------------------
// Tiled WMMA GEMM: C[M,NTOK] = A[M,K](f16) * B[K,NTOK](f16), fused epilogue.
//   cs != null  : conv/BN fold  v = (acc + cb[row]) * cs[row] + ct[row]
//   bias != null: v = acc + bias[row]
//   act: 0 none, 1 relu, 2 exact gelu
//   resid != null: v += resid[row*NTOK+col]   (applied AFTER activation)
//   Ch != null  : also store f16 copy (feeds the next GEMM)
// Block: 256 threads = 8 waves; tile 128x128, K-step 32.
// ---------------------------------------------------------------------------
__global__ __launch_bounds__(256) void gemm_wmma_kernel(
    const _Float16* __restrict__ A, const _Float16* __restrict__ Bm,
    float* __restrict__ C, _Float16* __restrict__ Ch,
    const float* __restrict__ bias,
    const float* __restrict__ cb, const float* __restrict__ cs,
    const float* __restrict__ ct,
    const float* __restrict__ resid,
    int M, int K, int act) {
  __shared__ __align__(16) _Float16 As[128 * 32];   // [m][kk]
  __shared__ __align__(16) _Float16 BsT[128 * 32];  // [n][kk]  (transposed!)

  const int tid  = threadIdx.x;
  const int lane = tid & 31;
  const int wave = tid >> 5;
  const int wm   = wave >> 1;   // 0..3 -> 32-row slab
  const int wn   = wave & 1;    // 0..1 -> 64-col slab
  const int bm0  = blockIdx.y * 128;
  const int bn0  = blockIdx.x * 128;

  v8f acc[2][4];
#pragma unroll
  for (int i = 0; i < 2; ++i)
#pragma unroll
    for (int j = 0; j < 4; ++j) acc[i][j] = zero_v8f();

  // B staging (transposing): thread owns column n and a 16-deep K slab.
  const int bt_n  = tid & 127;            // 0..127
  const int bt_k  = (tid >> 7) * 16;      // 0 or 16

#if HAVE_TDM
  // Tensor Data Mover: one 2D descriptor moves the whole 128x32 A tile.
  // (LDS aperture keeps the byte offset in the low 32 address bits.)
  const unsigned as_lds = (unsigned)(uintptr_t)(&As[0]);
  const unsigned long long abase =
      (unsigned long long)(uintptr_t)(A + (size_t)bm0 * K);
#else
  // Per-thread async global->LDS fallback: 16 halves (two 16B chunks).
  const int a_lin = tid * 16;
  const int a_m   = a_lin >> 5;     // 0..127
  const int a_k   = a_lin & 31;     // 0 or 16
  const unsigned a_lds = (unsigned)(uintptr_t)(As + a_m * 32 + a_k);
#endif

  for (int k0 = 0; k0 < K; k0 += 32) {
    // ---- A tile: TDM DMA global -> LDS (TENSORcnt), or async fallback ----
#if HAVE_TDM
    if (wave == 0) {
      const unsigned long long ga = abase + (unsigned long long)k0 * 2ull;
      v4u g0;
      g0[0] = 1u;                                  // count=1 (user D#)
      g0[1] = as_lds;                              // lds_addr
      g0[2] = (unsigned)(ga & 0xffffffffu);        // global_addr[31:0]
      g0[3] = (unsigned)((ga >> 32) & 0x01ffffffu) // global_addr[56:32]
              | (2u << 30);                        // type = 2 ("image")
      v8i_ g1;
      g1[0] = (int)(1u << 16);                     // data_size = 2 bytes
      g1[1] = (int)(((unsigned)K & 0xffffu) << 16);// tensor_dim0[15:0]
      g1[2] = (int)(((unsigned)K >> 16) | (128u << 16)); // dim0[31:16], dim1=128
      g1[3] = (int)(32u << 16);                    // tile_dim0 = 32
      g1[4] = (int)128u;                           // tile_dim1 = 128, tile_dim2=0
      g1[5] = (int)K;                              // tensor_dim0_stride[31:0]
      g1[6] = 0;                                   // stride0[47:32], stride1[15:0]
      g1[7] = 0;                                   // stride1[47:16]
      const v4i_ z4 = {0, 0, 0, 0};
      const v8i_ z8 = {0, 0, 0, 0, 0, 0, 0, 0};
      __builtin_amdgcn_tensor_load_to_lds(g0, g1, z4, z4, z8, 0);
      __builtin_amdgcn_s_wait_tensorcnt(0);
    }
#else
    {
      const _Float16* ga = A + (size_t)(bm0 + a_m) * K + (k0 + a_k);
      asm volatile("global_load_async_to_lds_b128 %0, %1, off"
                   :: "v"(a_lds), "v"(ga) : "memory");
      asm volatile("global_load_async_to_lds_b128 %0, %1, off"
                   :: "v"(a_lds + 16u), "v"(ga + 8) : "memory");
    }
#endif
    // ---- B tile: global column reads (wave-coalesced u16, offsets fold
    //      into immediates since NTOK is constant) -> LDS b128, transposed ----
    {
      const _Float16* gb = Bm + (size_t)(k0 + bt_k) * NTOK + (bn0 + bt_n);
      v8h lo, hi;
#pragma unroll
      for (int j = 0; j < 8; ++j) lo[j] = gb[(size_t)j * NTOK];
#pragma unroll
      for (int j = 0; j < 8; ++j) hi[j] = gb[(size_t)(8 + j) * NTOK];
      v8h* db = (v8h*)(BsT + bt_n * 32 + bt_k);
      db[0] = lo;
      db[1] = hi;
    }
    if (k0 + 32 < K) {  // global_prefetch_b8 for the next B K-slab
      __builtin_prefetch(Bm + (size_t)(k0 + 32 + bt_k) * NTOK + (bn0 + bt_n),
                         0, 3);
    }
#if !HAVE_TDM
    asm volatile("s_wait_asynccnt 0x0" ::: "memory");
#endif
    __syncthreads();

    v16h af[2], bf[4];
#pragma unroll
    for (int i = 0; i < 2; ++i)
      af[i] = load_a_frag32(As, wm * 32 + i * 16, lane);
#pragma unroll
    for (int j = 0; j < 4; ++j)
      bf[j] = load_b_frag32T(BsT, wn * 64 + j * 16, lane);
#pragma unroll
    for (int i = 0; i < 2; ++i)
#pragma unroll
      for (int j = 0; j < 4; ++j) acc[i][j] = wmma32(af[i], bf[j], acc[i][j]);
    __syncthreads();
  }

  // Epilogue. C/D layout: VGPR r -> M = r + 8*(lane>=16), N = lane&15.
#pragma unroll
  for (int i = 0; i < 2; ++i) {
#pragma unroll
    for (int j = 0; j < 4; ++j) {
#pragma unroll
      for (int r = 0; r < 8; ++r) {
        const int row = bm0 + wm * 32 + i * 16 + ((lane >> 4) & 1) * 8 + r;
        const int col = bn0 + wn * 64 + j * 16 + (lane & 15);
        float v = acc[i][j][r];
        if (cs)        v = (v + cb[row]) * cs[row] + ct[row];
        else if (bias) v += bias[row];
        if (act == 1)      v = fmaxf(v, 0.0f);
        else if (act == 2) v = 0.5f * v * (1.0f + erff(v * 0.70710678118f));
        const size_t off = (size_t)row * NTOK + col;
        if (resid) v += resid[off];
        C[off] = v;
        if (Ch) Ch[off] = (_Float16)v;
      }
    }
  }
}

// ---------------------------------------------------------------------------
// Flash attention.  qkv f16 feature-major [3*D, NTOK]; per wave: 16 queries,
// DH=32 (== one WMMA K), key chunks of 32, online softmax through LDS.
// grid = (L/128, H, B), block = 256 (8 waves).
// ---------------------------------------------------------------------------
__global__ __launch_bounds__(256) void attn_kernel(
    const _Float16* __restrict__ qkv, float* __restrict__ outf,
    _Float16* __restrict__ outh) {
  __shared__ __align__(16) float    sS[8][16][33];   // raw scores (padded)
  __shared__ __align__(16) _Float16 sP[8][16][32];   // exp-probs (A-frag src)
  __shared__ float sCorr[8][16];
  __shared__ float sLinv[8][16];

  const int tid  = threadIdx.x;
  const int lane = tid & 31;
  const int wave = tid >> 5;
  const int hi   = (lane >> 4) & 1;
  const int ln   = lane & 15;
  const int b    = blockIdx.z;
  const int h    = blockIdx.y;
  const int q0   = blockIdx.x * 128 + wave * 16;

  const _Float16* Qp = qkv + (size_t)(h * DH_) * NTOK + b * L_;
  const _Float16* Kp = qkv + (size_t)(D_ + h * DH_) * NTOK + b * L_;
  const _Float16* Vp = qkv + (size_t)(2 * D_ + h * DH_) * NTOK + b * L_;

  // Q fragment: A[m][k] = Q[dh=k, token=q0+m] (feature-major => transposed)
  const v16h aq = load_a_fragT_g(Qp + q0, NTOK, lane);

  v8f o[2];
  o[0] = zero_v8f();
  o[1] = zero_v8f();
  float mrow = -1e30f, lrow = 0.0f;

  for (int kc = 0; kc < L_; kc += 32) {
    // S = Q * K^T : B[k=dh][n=key] is exactly feature-major K
    const v16h bk0 = load_b_frag_g(Kp, NTOK, kc, lane);
    const v16h bk1 = load_b_frag_g(Kp, NTOK, kc + 16, lane);
    const v8f  s0  = wmma32(aq, bk0, zero_v8f());
    const v8f  s1  = wmma32(aq, bk1, zero_v8f());
#pragma unroll
    for (int r = 0; r < 8; ++r) {
      const int m = r + hi * 8;
      sS[wave][m][ln]      = s0[r] * SOFT_SCALE;
      sS[wave][m][16 + ln] = s1[r] * SOFT_SCALE;
    }
    __syncthreads();

    if (lane < 16) {  // one lane per query row: online softmax stats
      float mx = -1e30f;
#pragma unroll 4
      for (int j = 0; j < 32; ++j) mx = fmaxf(mx, sS[wave][lane][j]);
      const float mnew = fmaxf(mrow, mx);
      const float corr = __expf(mrow - mnew);
      float sum = 0.0f;
#pragma unroll 4
      for (int j = 0; j < 32; ++j) {
        const float p = __expf(sS[wave][lane][j] - mnew);
        sP[wave][lane][j] = (_Float16)p;
        sum += p;
      }
      lrow = lrow * corr + sum;
      mrow = mnew;
      sCorr[wave][lane] = corr;
    }
    __syncthreads();

#pragma unroll
    for (int t = 0; t < 2; ++t)
#pragma unroll
      for (int r = 0; r < 8; ++r) o[t][r] *= sCorr[wave][r + hi * 8];

    // O += P * V : P from LDS as vectorized A fragment; V per-lane contiguous
    const v16h ap  = load_a_frag32(&sP[wave][0][0], 0, lane);
    const v16h bv0 = load_b_fragT_g(Vp + kc, NTOK, 0, lane);
    const v16h bv1 = load_b_fragT_g(Vp + kc, NTOK, 16, lane);
    o[0] = wmma32(ap, bv0, o[0]);
    o[1] = wmma32(ap, bv1, o[1]);
    __syncthreads();
  }

  if (lane < 16) sLinv[wave][lane] = 1.0f / lrow;
  __syncthreads();

#pragma unroll
  for (int t = 0; t < 2; ++t)
#pragma unroll
    for (int r = 0; r < 8; ++r) {
      const int m  = r + hi * 8;
      const int dh = t * 16 + ln;
      const float v = o[t][r] * sLinv[wave][m];
      const size_t off = (size_t)(h * DH_ + dh) * NTOK + (size_t)b * L_ + q0 + m;
      outf[off] = v;
      outh[off] = (_Float16)v;
    }
}

// ---------------------------------------------------------------------------
// Elementwise / small kernels
// ---------------------------------------------------------------------------
__global__ void f32_to_f16_kernel(const float* __restrict__ in,
                                  _Float16* __restrict__ out, int n) {
  const int i = blockIdx.x * 256 + threadIdx.x;
  if (i < n) out[i] = (_Float16)in[i];
}

// LayerNorm across features (feature-major [D, NTOK]); one token per thread.
__global__ __launch_bounds__(256) void ln_kernel(
    const float* __restrict__ x, const float* __restrict__ g,
    const float* __restrict__ bb, float* __restrict__ outf,
    _Float16* __restrict__ outh) {
  const int n = blockIdx.x * 256 + threadIdx.x;
  if (n >= NTOK) return;
  float sum = 0.0f, sq = 0.0f;
  for (int d = 0; d < D_; ++d) {
    const float v = x[(size_t)d * NTOK + n];
    sum += v;
    sq  += v * v;
  }
  const float mu   = sum * (1.0f / D_);
  const float var  = sq * (1.0f / D_) - mu * mu;
  const float rstd = rsqrtf(var + 1e-5f);
  for (int d = 0; d < D_; ++d) {
    const size_t off = (size_t)d * NTOK + n;
    const float v = (x[off] - mu) * rstd * g[d] + bb[d];
    outf[off] = v;
    outh[off] = (_Float16)v;
  }
}

__global__ void add_pe_kernel(float* __restrict__ t, _Float16* __restrict__ th) {
  const int idx = blockIdx.x * 256 + threadIdx.x;
  if (idx >= D_ * NTOK) return;
  const int d = idx / NTOK;
  const int n = idx - d * NTOK;
  const int l = n & (L_ - 1);
  const float div = __expf((float)((d >> 1) * 2) * (-9.210340371976184f / (float)D_));
  const float ang = (float)l * div;
  const float pe  = (d & 1) ? __cosf(ang) : __sinf(ang);
  const float v = t[idx] + pe;
  t[idx]  = v;
  th[idx] = (_Float16)v;
}

// im2col for K=11 pad=5 conv, feature-major in [F, NTOK] -> col f16 [CKK, NTOK]
__global__ void im2col_kernel(const float* __restrict__ x,
                              _Float16* __restrict__ col) {
  const int idx = blockIdx.x * 256 + threadIdx.x;
  if (idx >= CKK * NTOK) return;
  const int r = idx / NTOK;
  const int n = idx - r * NTOK;
  const int c = r / KW;
  const int k = r - c * KW;
  const int b = n >> 11;
  const int l = n & (L_ - 1);
  const int ls = l + k - PAD_;
  const float v = (ls >= 0 && ls < L_) ? x[(size_t)c * NTOK + (b << 11) + ls] : 0.0f;
  col[idx] = (_Float16)v;
}

// Input embedding conv (CIN=1) + BN fold + relu, direct.
__global__ void emb_conv_kernel(const float* __restrict__ x,
                                const float* __restrict__ w,
                                const float* __restrict__ eb,
                                const float* __restrict__ es,
                                const float* __restrict__ et,
                                float* __restrict__ out) {
  const int idx = blockIdx.x * 256 + threadIdx.x;
  if (idx >= F_ * NTOK) return;
  const int f = idx / NTOK;
  const int n = idx - f * NTOK;
  const int b = n >> 11;
  const int l = n & (L_ - 1);
  float sum = 0.0f;
#pragma unroll
  for (int k = 0; k < KW; ++k) {
    const int ls = l + k - PAD_;
    if (ls >= 0 && ls < L_) sum += w[f * KW + k] * x[(b << 11) + ls];
  }
  const float v = (sum + eb[f]) * es[f] + et[f];
  out[idx] = fmaxf(v, 0.0f);
}

// Output conv (1 channel) + sigmoid, direct.
__global__ void out_conv_kernel(const float* __restrict__ h,
                                const float* __restrict__ w,
                                const float* __restrict__ ob,
                                float* __restrict__ out) {
  const int n = blockIdx.x * 256 + threadIdx.x;
  if (n >= NTOK) return;
  const int b = n >> 11;
  const int l = n & (L_ - 1);
  float sum = 0.0f;
  for (int c = 0; c < F_; ++c) {
#pragma unroll
    for (int k = 0; k < KW; ++k) {
      const int ls = l + k - PAD_;
      if (ls >= 0 && ls < L_)
        sum += w[c * KW + k] * h[(size_t)c * NTOK + (b << 11) + ls];
    }
  }
  out[n] = 1.0f / (1.0f + __expf(-(sum + ob[0])));
}

// ---------------------------------------------------------------------------
// Host orchestration
// ---------------------------------------------------------------------------
extern "C" void kernel_launch(void* const* d_in, const int* in_sizes, int n_in,
                              void* d_out, int out_size, void* d_ws,
                              size_t ws_size, hipStream_t stream) {
  (void)in_sizes; (void)n_in; (void)out_size; (void)ws_size;

  const float* x       = (const float*)d_in[0];
  const float* emb_w   = (const float*)d_in[1];
  const float* emb_b   = (const float*)d_in[2];
  const float* emb_s   = (const float*)d_in[3];
  const float* emb_t   = (const float*)d_in[4];
  const float* renc_w1 = (const float*)d_in[5];
  const float* renc_b1 = (const float*)d_in[6];
  const float* renc_s1 = (const float*)d_in[7];
  const float* renc_t1 = (const float*)d_in[8];
  const float* renc_w2 = (const float*)d_in[9];
  const float* renc_b2 = (const float*)d_in[10];
  const float* renc_s2 = (const float*)d_in[11];
  const float* renc_t2 = (const float*)d_in[12];
  const float* ln1_g   = (const float*)d_in[13];
  const float* ln1_b   = (const float*)d_in[14];
  const float* qkv_w   = (const float*)d_in[15];
  const float* proj_w  = (const float*)d_in[16];
  const float* proj_b  = (const float*)d_in[17];
  const float* ln2_g   = (const float*)d_in[18];
  const float* ln2_b   = (const float*)d_in[19];
  const float* fc1_w   = (const float*)d_in[20];
  const float* fc1_b   = (const float*)d_in[21];
  const float* fc2_w   = (const float*)d_in[22];
  const float* fc2_b   = (const float*)d_in[23];
  const float* rdec_w1 = (const float*)d_in[24];
  const float* rdec_b1 = (const float*)d_in[25];
  const float* rdec_s1 = (const float*)d_in[26];
  const float* rdec_t1 = (const float*)d_in[27];
  const float* rdec_w2 = (const float*)d_in[28];
  const float* rdec_b2 = (const float*)d_in[29];
  const float* rdec_s2 = (const float*)d_in[30];
  const float* rdec_t2 = (const float*)d_in[31];
  const float* out_w   = (const float*)d_in[32];
  const float* out_b   = (const float*)d_in[33];

  // bump allocator on the workspace
  char* wp = (char*)d_ws;
  auto alloc = [&](size_t bytes) -> void* {
    void* p = (void*)wp;
    wp += (bytes + 255) & ~(size_t)255;
    return p;
  };

  const size_t FFK = (size_t)F_ * F_ * KW;  // 720896 elems per conv weight

  // f16 weight shadows
  _Float16* w_renc1 = (_Float16*)alloc(R_ * FFK * 2);
  _Float16* w_renc2 = (_Float16*)alloc(R_ * FFK * 2);
  _Float16* w_rdec1 = (_Float16*)alloc(R_ * FFK * 2);
  _Float16* w_rdec2 = (_Float16*)alloc(R_ * FFK * 2);
  _Float16* w_qkv   = (_Float16*)alloc((size_t)R_ * 3 * D_ * D_ * 2);
  _Float16* w_proj  = (_Float16*)alloc((size_t)R_ * D_ * D_ * 2);
  _Float16* w_fc1   = (_Float16*)alloc((size_t)R_ * 4 * D_ * D_ * 2);
  _Float16* w_fc2   = (_Float16*)alloc((size_t)R_ * 4 * D_ * D_ * 2);

  // activation buffers (feature-major [C, NTOK])
  float*    actA   = (float*)alloc((size_t)F_ * NTOK * 4);
  float*    actB   = (float*)alloc((size_t)F_ * NTOK * 4);
  _Float16* colbuf = (_Float16*)alloc((size_t)CKK * NTOK * 2);
  _Float16* t_h    = (_Float16*)alloc((size_t)D_ * NTOK * 2);
  float*    qkv_f  = (float*)alloc((size_t)3 * D_ * NTOK * 4);
  _Float16* qkv_h  = (_Float16*)alloc((size_t)3 * D_ * NTOK * 2);
  float*    attn_f = (float*)alloc((size_t)D_ * NTOK * 4);
  _Float16* attn_h = (_Float16*)alloc((size_t)D_ * NTOK * 2);
  float*    preln  = (float*)alloc((size_t)D_ * NTOK * 4);
  float*    y1_f   = (float*)alloc((size_t)4 * D_ * NTOK * 4);
  _Float16* y1_h   = (_Float16*)alloc((size_t)4 * D_ * NTOK * 2);

  auto cvt = [&](const float* src, _Float16* dst, int n) {
    f32_to_f16_kernel<<<(n + 255) / 256, 256, 0, stream>>>(src, dst, n);
  };
  auto gemm = [&](const _Float16* A, const _Float16* Bm, float* C, _Float16* Ch,
                  const float* bias, const float* cb, const float* cs,
                  const float* ct, const float* resid, int M, int K, int act) {
    dim3 g(NTOK / 128, M / 128);
    gemm_wmma_kernel<<<g, 256, 0, stream>>>(A, Bm, C, Ch, bias, cb, cs, ct,
                                            resid, M, K, act);
  };
  auto im2col = [&](const float* src) {
    im2col_kernel<<<(CKK * NTOK + 255) / 256, 256, 0, stream>>>(src, colbuf);
  };

  // 0) weights -> f16
  cvt(renc_w1, w_renc1, (int)(R_ * FFK));
  cvt(renc_w2, w_renc2, (int)(R_ * FFK));
  cvt(rdec_w1, w_rdec1, (int)(R_ * FFK));
  cvt(rdec_w2, w_rdec2, (int)(R_ * FFK));
  cvt(qkv_w,  w_qkv,  R_ * 3 * D_ * D_);
  cvt(proj_w, w_proj, R_ * D_ * D_);
  cvt(fc1_w,  w_fc1,  R_ * 4 * D_ * D_);
  cvt(fc2_w,  w_fc2,  R_ * 4 * D_ * D_);

  // 1) input embedding conv (CIN=1, direct) -> actA [F, NTOK]
  emb_conv_kernel<<<(F_ * NTOK + 255) / 256, 256, 0, stream>>>(
      x, emb_w, emb_b, emb_s, emb_t, actA);

  // 2) encoder ResCNN blocks (implicit-GEMM conv via im2col + WMMA)
  for (int i = 0; i < R_; ++i) {
    im2col(actA);
    gemm(w_renc1 + i * FFK, colbuf, actB, nullptr, nullptr,
         renc_b1 + i * F_, renc_s1 + i * F_, renc_t1 + i * F_, nullptr,
         F_, CKK, /*relu*/ 1);
    im2col(actB);
    gemm(w_renc2 + i * FFK, colbuf, actA, nullptr, nullptr,
         renc_b2 + i * F_, renc_s2 + i * F_, renc_t2 + i * F_, /*resid*/ actA,
         F_, CKK, /*relu*/ 1);
  }

  // 3) + positional encoding (also produces f16 copy for the qkv GEMM)
  add_pe_kernel<<<(D_ * NTOK + 255) / 256, 256, 0, stream>>>(actA, t_h);

  // 4) post-norm transformer blocks
  for (int i = 0; i < R_; ++i) {
    gemm(w_qkv + (size_t)i * 3 * D_ * D_, t_h, qkv_f, qkv_h, nullptr, nullptr,
         nullptr, nullptr, nullptr, 3 * D_, D_, 0);
    attn_kernel<<<dim3(L_ / 128, H_, B_), 256, 0, stream>>>(qkv_h, attn_f,
                                                            attn_h);
    gemm(w_proj + (size_t)i * D_ * D_, attn_h, preln, nullptr,
         proj_b + i * D_, nullptr, nullptr, nullptr, /*resid*/ actA, D_, D_, 0);
    ln_kernel<<<NTOK / 256, 256, 0, stream>>>(preln, ln1_g + i * D_,
                                              ln1_b + i * D_, actA, t_h);
    gemm(w_fc1 + (size_t)i * 4 * D_ * D_, t_h, y1_f, y1_h, fc1_b + i * 4 * D_,
         nullptr, nullptr, nullptr, nullptr, 4 * D_, D_, /*gelu*/ 2);
    gemm(w_fc2 + (size_t)i * 4 * D_ * D_, y1_h, preln, nullptr, fc2_b + i * D_,
         nullptr, nullptr, nullptr, /*resid*/ actA, D_, 4 * D_, 0);
    ln_kernel<<<NTOK / 256, 256, 0, stream>>>(preln, ln2_g + i * D_,
                                              ln2_b + i * D_, actA, t_h);
  }

  // 5) decoder ResCNN blocks
  for (int i = 0; i < R_; ++i) {
    im2col(actA);
    gemm(w_rdec1 + i * FFK, colbuf, actB, nullptr, nullptr,
         rdec_b1 + i * F_, rdec_s1 + i * F_, rdec_t1 + i * F_, nullptr,
         F_, CKK, 1);
    im2col(actB);
    gemm(w_rdec2 + i * FFK, colbuf, actA, nullptr, nullptr,
         rdec_b2 + i * F_, rdec_s2 + i * F_, rdec_t2 + i * F_, actA,
         F_, CKK, 1);
  }

  // 6) output conv + sigmoid -> d_out [B,1,L]
  out_conv_kernel<<<NTOK / 256, 256, 0, stream>>>(actA, out_w, out_b,
                                                  (float*)d_out);
}
